// HQQLinear_19851338842900
// MI455X (gfx1250) — compile-verified, matrix operand order checked by reading, not run on hardware
//
#include <hip/hip_runtime.h>
#include <hip/hip_bf16.h>
#include <stdint.h>

// ---------------------------------------------------------------------------
// HQQ 4-bit dequant + GEMM for MI455X (gfx1250, wave32, WMMA bf16 path)
//   out[4096(M) x 4096(N)] = x[M x K=4096] * W_est[N x K]^T + bias
//   1) dequant W_q (packed 4-bit) -> bf16 W_est in workspace (32 MB)
//   2) convert x f32 -> bf16 in workspace (32 MB)
//   3) GEMM: 128x256 block tile, 8 waves (2M x 4N), 64x64 wave tile,
//      double-buffered fragment prefetch, v_wmma_f32_16x16x32_bf16.
//      SGPR base + 32-bit VGPR offsets (saddr addressing) to minimize
//      per-iteration address VALU and WMMA->VALU hazard NOPs.
// ---------------------------------------------------------------------------

typedef __attribute__((ext_vector_type(16))) __bf16 v16bf;
typedef __attribute__((ext_vector_type(4)))  __bf16 v4bf;
typedef __attribute__((ext_vector_type(8)))  float  v8f;

#define IN_F   4096
#define OUT_F  4096
#define MTOT   4096      // 8*512
#define NGROUP 262144    // IN_F*OUT_F/64

// ---------------- dequant: W_q[32, 262144] int32 (byte codes) -> bf16 [4096,4096]
__global__ __launch_bounds__(256)
void hqq_dequant_kernel(const int* __restrict__ Wq,
                        const float* __restrict__ scale,
                        const float* __restrict__ zero,
                        __bf16* __restrict__ Wb)
{
    int t = blockIdx.x * 256 + threadIdx.x;          // 0 .. 4194303
    int e = t << 2;                                  // element index into [4096*4096]
    int o = e >> 12;                                 // row (out feature)
    int i = e & 4095;                                // col (in feature), multiple of 4
    int g = o >> 6;                                  // 0..63
    int n = ((o & 63) << 12) + i;                    // group id, multiple of 4
    int row = (g < 32) ? g : (g - 32);

    const int4*   pq = (const int4*)(Wq + (size_t)row * NGROUP + n);
    const float4* ps = (const float4*)(scale + n);
    const float4* pz = (const float4*)(zero  + n);
    int4   q = *pq;
    float4 s = *ps;
    float4 z = *pz;

    bool hi = (g < 32);
    float c0 = (float)(hi ? ((q.x >> 4) & 0xF) : (q.x & 0xF));
    float c1 = (float)(hi ? ((q.y >> 4) & 0xF) : (q.y & 0xF));
    float c2 = (float)(hi ? ((q.z >> 4) & 0xF) : (q.z & 0xF));
    float c3 = (float)(hi ? ((q.w >> 4) & 0xF) : (q.w & 0xF));

    v4bf r;
    r.x = (__bf16)((c0 - z.x) * s.x);
    r.y = (__bf16)((c1 - z.y) * s.y);
    r.z = (__bf16)((c2 - z.z) * s.z);
    r.w = (__bf16)((c3 - z.w) * s.w);
    *(v4bf*)(Wb + e) = r;
}

// ---------------- x: f32 -> bf16
__global__ __launch_bounds__(256)
void cvt_bf16_kernel(const float* __restrict__ x, __bf16* __restrict__ xb)
{
    int t = blockIdx.x * 256 + threadIdx.x;
    int e = t << 2;
    float4 v = *(const float4*)(x + e);
    v4bf r;
    r.x = (__bf16)v.x; r.y = (__bf16)v.y; r.z = (__bf16)v.z; r.w = (__bf16)v.w;
    *(v4bf*)(xb + e) = r;
}

// ---------------- GEMM
__global__ __launch_bounds__(256)
void wmma_gemm_kernel(const __bf16* __restrict__ A,   // [MTOT, IN_F] bf16 (x)
                      const __bf16* __restrict__ B,   // [OUT_F, IN_F] bf16 (W_est)
                      const float*  __restrict__ bias,
                      float* __restrict__ C)          // [MTOT, OUT_F] f32
{
    const int lane = threadIdx.x & 31;
    const int wave = threadIdx.x >> 5;
    const int wm   = wave >> 2;        // 0..1
    const int wn   = wave & 3;         // 0..3

    const int m_base = blockIdx.y * 128 + wm * 64;   // wave's 64-row band
    const int n_base = blockIdx.x * 256 + wn * 64;   // wave's 64-col band

    const int lr = lane & 15;          // row/col within 16-tile
    const int kh = (lane >> 4) * 16;   // K half selected by lane group

    // per-lane 32-bit element offsets; loads become saddr + voffset form
    int aoff[4];
    int boff[4];
#pragma unroll
    for (int mi = 0; mi < 4; ++mi)
        aoff[mi] = (m_base + mi * 16 + lr) * IN_F + kh;
#pragma unroll
    for (int ni = 0; ni < 4; ++ni)
        boff[ni] = (n_base + ni * 16 + lr) * IN_F + kh;

    v8f acc[4][4];
#pragma unroll
    for (int mi = 0; mi < 4; ++mi)
#pragma unroll
        for (int ni = 0; ni < 4; ++ni)
            acc[mi][ni] = (v8f){0.f, 0.f, 0.f, 0.f, 0.f, 0.f, 0.f, 0.f};

    v16bf a0[4], b0[4], a1[4], b1[4];

    // preload buffer 0 at k=0
#pragma unroll
    for (int mi = 0; mi < 4; ++mi) a0[mi] = *(const v16bf*)(A + aoff[mi]);
#pragma unroll
    for (int ni = 0; ni < 4; ++ni) b0[ni] = *(const v16bf*)(B + boff[ni]);

    for (int k0 = 0; k0 < IN_F; k0 += 64) {
        // prefetch buffer 1 at k0+32 (always in range: k0 <= K-64)
        const int k1 = k0 + 32;
#pragma unroll
        for (int mi = 0; mi < 4; ++mi) a1[mi] = *(const v16bf*)(A + (aoff[mi] + k1));
#pragma unroll
        for (int ni = 0; ni < 4; ++ni) b1[ni] = *(const v16bf*)(B + (boff[ni] + k1));

        // WMMAs on buffer 0 (k0)
#pragma unroll
        for (int mi = 0; mi < 4; ++mi)
#pragma unroll
            for (int ni = 0; ni < 4; ++ni)
                acc[mi][ni] = __builtin_amdgcn_wmma_f32_16x16x32_bf16(
                    false, a0[mi], false, b0[ni],
                    (short)0, acc[mi][ni], false, false);

        // prefetch buffer 0 at k0+64; wraps to a harmless dummy k=0 on the
        // final iteration (K is a power of two)
        const int k2 = (k0 + 64) & (IN_F - 1);
#pragma unroll
        for (int mi = 0; mi < 4; ++mi) a0[mi] = *(const v16bf*)(A + (aoff[mi] + k2));
#pragma unroll
        for (int ni = 0; ni < 4; ++ni) b0[ni] = *(const v16bf*)(B + (boff[ni] + k2));

        // WMMAs on buffer 1 (k0+32)
#pragma unroll
        for (int mi = 0; mi < 4; ++mi)
#pragma unroll
            for (int ni = 0; ni < 4; ++ni)
                acc[mi][ni] = __builtin_amdgcn_wmma_f32_16x16x32_bf16(
                    false, a1[mi], false, b1[ni],
                    (short)0, acc[mi][ni], false, false);
    }

    // epilogue: C VGPR j holds M = j (+8 for lanes 16..31), N = lane&15
    const int mo = (lane >> 4) << 3;   // 0 or 8
#pragma unroll
    for (int mi = 0; mi < 4; ++mi) {
#pragma unroll
        for (int ni = 0; ni < 4; ++ni) {
            const int col = n_base + ni * 16 + lr;
            const float bv = bias[col];
            const int rbase = m_base + mi * 16 + mo;
#pragma unroll
            for (int j = 0; j < 8; ++j)
                C[(size_t)(rbase + j) * OUT_F + col] = acc[mi][ni][j] + bv;
        }
    }
}

extern "C" void kernel_launch(void* const* d_in, const int* in_sizes, int n_in,
                              void* d_out, int out_size, void* d_ws, size_t ws_size,
                              hipStream_t stream)
{
    const float* x     = (const float*)d_in[0];
    const int*   Wq    = (const int*)  d_in[1];
    const float* scale = (const float*)d_in[2];
    const float* zero  = (const float*)d_in[3];
    const float* bias  = (const float*)d_in[4];
    float* out = (float*)d_out;

    __bf16* Wb = (__bf16*)d_ws;                       // 4096*4096 bf16 = 32 MB
    __bf16* Xb = Wb + (size_t)OUT_F * IN_F;           // 4096*4096 bf16 = 32 MB

    // 1) dequantize weights -> bf16 (16.7M elems, 4 per thread)
    hqq_dequant_kernel<<<(OUT_F * IN_F) / (256 * 4), 256, 0, stream>>>(Wq, scale, zero, Wb);
    // 2) convert activations -> bf16
    cvt_bf16_kernel<<<(MTOT * IN_F) / (256 * 4), 256, 0, stream>>>(x, Xb);
    // 3) WMMA GEMM with fused bias
    dim3 grid(OUT_F / 256, MTOT / 128);
    wmma_gemm_kernel<<<grid, 256, 0, stream>>>(Xb, Wb, bias, out);
}
